// PCENFrontend_30288109372166
// MI455X (gfx1250) — matrix operand order repaired and checked by reading, not compiled, not run on hardware
//
#include <hip/hip_runtime.h>
#include <hip/hip_bf16.h>
#include <math.h>

// ---------------- problem constants ----------------
#define LSAMP   640000      // samples per batch row
#define NBATCH  32
#define NFFT    1024
#define HOP     256
#define TFRAMES 2501        // 1 + (640000+1024-1024)/256
#define FTILES  157         // ceil(2501/16)
#define NMELS   128
#define NBINS   513
#define NT1     33          // freq tiles of 16 (528 >= 513)
#define KT1     32          // 1024 / 32
#define NT2     8           // 128 / 16
#define KT2     17          // 544 / 32
#define SCONST  0.04f
#define EPSC    1e-6f

// fragment sizes (elements): one WMMA B fragment tile = 32 lanes * 16 bf16
#define SEG      512
#define WR_ELEMS (KT1 * NT1 * SEG)   // 540672
#define FB_ELEMS (KT2 * NT2 * SEG)   // 69632
#define WR_BYTES ((size_t)WR_ELEMS * 2)
#define FB_BYTES ((size_t)FB_ELEMS * 2)
#define E_OFF    (2 * WR_BYTES + FB_BYTES)          // 2,301,952 B (256B aligned)
// total ws need: E_OFF + 32*128*2501*4 = ~43.3 MB

typedef __attribute__((ext_vector_type(16))) __bf16 v16bf;
typedef __attribute__((ext_vector_type(8)))  float  v8f;

__device__ __forceinline__ float fast_pow(float x, float y) {
    return __expf(y * __logf(x));
}

// ---------------------------------------------------------------------------
// Kernel 0: generate DFT (cos / -sin) matrices and mel filterbank, already
// swizzled into this code's WMMA bf16 B-fragment lane layout (contiguous k, as A):
//   fragment tile (kt,nt): element [lane][e]  ->  K = kt*32 + e + 16*(lane>>4)
//                                                N = nt*16 + (lane&15)
// ---------------------------------------------------------------------------
__global__ void gen_frags(const float* __restrict__ fb,
                          __bf16* __restrict__ wr,
                          __bf16* __restrict__ wi,
                          __bf16* __restrict__ fbf) {
    int idx = blockIdx.x * blockDim.x + threadIdx.x;
    const float TWO_PI = 6.28318530717958647692f;
    if (idx < 2 * WR_ELEMS) {
        bool imag = idx >= WR_ELEMS;
        int j    = imag ? idx - WR_ELEMS : idx;
        int ti   = j / SEG;
        int w    = j % SEG;
        int lane = w >> 4;
        int e    = w & 15;
        int kt   = ti / NT1;
        int nt   = ti % NT1;
        int k = kt * 32 + e + 16 * (lane >> 4);   // sample index 0..1023
        int f = nt * 16 + (lane & 15);            // freq index 0..527
        float v = 0.f;
        if (f <= NBINS - 1) {
            int p = (k * f) & (NFFT - 1);         // phase mod N for accuracy
            float theta = -TWO_PI * (float)p * (1.0f / (float)NFFT);
            v = imag ? __sinf(theta) : __cosf(theta);
        }
        if (imag) wi[j] = (__bf16)v; else wr[j] = (__bf16)v;
    } else if (idx < 2 * WR_ELEMS + FB_ELEMS) {
        int j    = idx - 2 * WR_ELEMS;
        int ti   = j / SEG;
        int w    = j % SEG;
        int lane = w >> 4;
        int e    = w & 15;
        int kt   = ti / NT2;
        int nt   = ti % NT2;
        int k = kt * 32 + e + 16 * (lane >> 4);   // freq index 0..543
        int m = nt * 16 + (lane & 15);            // mel index 0..127
        float v = (k < NBINS) ? fb[(size_t)k * NMELS + m] : 0.f;
        fbf[j] = (__bf16)v;
    }
}

// ---------------------------------------------------------------------------
// Kernel 1: fused windowed-DFT + magnitude + mel projection via bf16 WMMA.
// One block = (frame tile of 16, batch). LDS strides padded vs bank conflicts.
// ---------------------------------------------------------------------------
#define FR_STRIDE 1032   // bf16 elems per frame row in LDS (1024 used)
#define MG_STRIDE 552    // bf16 elems per mag row in LDS (544 used)

__global__ __launch_bounds__(128)
void mel_kernel(const float* __restrict__ x,
                const float* __restrict__ window,
                const __bf16* __restrict__ wr,
                const __bf16* __restrict__ wi,
                const __bf16* __restrict__ fbf,
                float* __restrict__ E) {
    __shared__ __bf16 sFrames[16 * FR_STRIDE];   // 33,024 B
    __shared__ __bf16 sMag[16 * MG_STRIDE];      // 17,664 B

    const int ft   = blockIdx.x;
    const int b    = blockIdx.y;
    const int tid  = threadIdx.x;
    const int lane = tid & 31;
    const int wv   = tid >> 5;          // wave id 0..3 (wave32)
    const int half = lane >> 4;
    const int Mrow = lane & 15;

    // --- stage frames into LDS (windowed, reflect-padded, bf16) ---
    const float* xb = x + (size_t)b * LSAMP;
    for (int i = tid; i < 16 * 1024; i += 128) {
        int fl = i >> 10;
        int k  = i & 1023;
        int t  = ft * 16 + fl;
        float v = 0.f;
        if (t < TFRAMES) {
            int pos = t * HOP + k - (NFFT / 2);  // reflect padding
            if (pos < 0) pos = -pos;
            else if (pos >= LSAMP) pos = 2 * LSAMP - 2 - pos;
            v = xb[pos] * window[k];
        }
        sFrames[fl * FR_STRIDE + k] = (__bf16)v;
    }
    for (int i = tid; i < 16 * MG_STRIDE; i += 128) sMag[i] = (__bf16)0.f;
    __syncthreads();

    // --- stage 1: frames(16x1024) x [Wr|Wi](1024x528) -> |X| in LDS ---
    for (int nt = wv; nt < NT1; nt += 4) {
        v8f accR = {};
        v8f accI = {};
        for (int kt = 0; kt < KT1; ++kt) {
            v16bf a;
#pragma unroll
            for (int e = 0; e < 16; ++e) {
                int K = e + half * 16 + kt * 32;  // A-frag map
                a[e] = sFrames[Mrow * FR_STRIDE + K];
            }
            const v16bf br = *(const v16bf*)(wr + ((size_t)(kt * NT1 + nt) * 32 + lane) * 16);
            const v16bf bi = *(const v16bf*)(wi + ((size_t)(kt * NT1 + nt) * 32 + lane) * 16);
            accR = __builtin_amdgcn_wmma_f32_16x16x32_bf16(false, a, false, br,
                                                           (short)0, accR, false, false);
            accI = __builtin_amdgcn_wmma_f32_16x16x32_bf16(false, a, false, bi,
                                                           (short)0, accI, false, false);
        }
#pragma unroll
        for (int e = 0; e < 8; ++e) {                 // D layout: M=e+8*half, N=lane&15
            int row = e + half * 8;
            int col = nt * 16 + (lane & 15);
            float re = accR[e], im = accI[e];
            sMag[row * MG_STRIDE + col] = (__bf16)__fsqrt_rn(re * re + im * im);
        }
    }
    __syncthreads();

    // --- stage 2: |X|(16x544) x fb(544x128) -> E tile (f32) ---
    for (int nt = wv; nt < NT2; nt += 4) {
        v8f acc = {};
        for (int kt = 0; kt < KT2; ++kt) {
            v16bf a;
#pragma unroll
            for (int e = 0; e < 16; ++e) {
                int K = e + half * 16 + kt * 32;
                a[e] = sMag[Mrow * MG_STRIDE + K];
            }
            const v16bf bb = *(const v16bf*)(fbf + ((size_t)(kt * NT2 + nt) * 32 + lane) * 16);
            acc = __builtin_amdgcn_wmma_f32_16x16x32_bf16(false, a, false, bb,
                                                          (short)0, acc, false, false);
        }
#pragma unroll
        for (int e = 0; e < 8; ++e) {
            int t_idx = ft * 16 + e + half * 8;
            int m     = nt * 16 + (lane & 15);
            if (t_idx < TFRAMES)
                E[((size_t)b * NMELS + m) * TFRAMES + t_idx] = acc[e];
        }
    }
}

// ---------------------------------------------------------------------------
// Kernel 2: PCEN. One wave32 per (b,m) row; chunked inclusive scan of the
// linear recurrence M[t] = a*M[t-1] + b via lane shuffles, carry across chunks.
// ---------------------------------------------------------------------------
__global__ __launch_bounds__(256)
void pcen_kernel(const float* __restrict__ E,
                 const float* __restrict__ alpha,
                 const float* __restrict__ delta,
                 const float* __restrict__ r,
                 float* __restrict__ out) {
    const int lane = threadIdx.x & 31;
    const int row  = blockIdx.x * 8 + (threadIdx.x >> 5);   // 4096 rows
    if (row >= NBATCH * NMELS) return;
    const int m = row & (NMELS - 1);
    const float al = alpha[m];
    const float de = delta[m];
    const float rr = r[m];
    const float dr = fast_pow(de, rr);

    const float* Erow = E + (size_t)row * TFRAMES;
    float*       Orow = out + (size_t)row * TFRAMES;

    float Mcarry = 0.f;
    for (int t0 = 0; t0 < TFRAMES; t0 += 32) {
        int t = t0 + lane;
        bool valid = (t < TFRAMES);
        float e  = valid ? Erow[t] : 0.f;
        float a  = valid ? ((t == 0) ? 1.0f : (1.0f - SCONST)) : 1.0f;
        float bq = valid ? ((t == 0) ? e : SCONST * e) : 0.0f;
        // inclusive scan over (a,b) with (a2,b2)o(a1,b1) = (a1*a2, a2*b1+b2)
#pragma unroll
        for (int off = 1; off < 32; off <<= 1) {
            float ap = __shfl_up(a, (unsigned)off, 32);
            float bp = __shfl_up(bq, (unsigned)off, 32);
            if (lane >= off) { bq = a * bp + bq; a = a * ap; }
        }
        float Mv = a * Mcarry + bq;
        if (valid) {
            float smooth = fast_pow(EPSC + Mv, al);
            Orow[t] = fast_pow(e / smooth + de, rr) - dr;
        }
        Mcarry = __shfl(Mv, 31, 32);
    }
}

// ---------------------------------------------------------------------------
extern "C" void kernel_launch(void* const* d_in, const int* in_sizes, int n_in,
                              void* d_out, int out_size, void* d_ws, size_t ws_size,
                              hipStream_t stream) {
    const float* x      = (const float*)d_in[0];   // (32, 640000)
    const float* alpha  = (const float*)d_in[1];   // (128, 1)
    const float* delta  = (const float*)d_in[2];   // (128, 1)
    const float* rr     = (const float*)d_in[3];   // (128, 1)
    const float* fb     = (const float*)d_in[4];   // (513, 128)
    const float* window = (const float*)d_in[5];   // (1024,)
    float* out = (float*)d_out;                    // (32, 1, 128, 2501) f32

    char* ws = (char*)d_ws;
    __bf16* wr  = (__bf16*)(ws);
    __bf16* wi  = (__bf16*)(ws + WR_BYTES);
    __bf16* fbf = (__bf16*)(ws + 2 * WR_BYTES);
    float*  E   = (float*)(ws + E_OFF);

    // 1) build WMMA-fragment-layout DFT + mel matrices
    {
        int total = 2 * WR_ELEMS + FB_ELEMS;
        gen_frags<<<(total + 255) / 256, 256, 0, stream>>>(fb, wr, wi, fbf);
    }
    // 2) fused windowed-DFT |.| + mel projection (bf16 WMMA, f32 accum)
    {
        dim3 grid(FTILES, NBATCH);
        mel_kernel<<<grid, 128, 0, stream>>>(x, window, wr, wi, fbf, E);
    }
    // 3) PCEN temporal scan + pointwise
    {
        pcen_kernel<<<(NBATCH * NMELS) / 8, 256, 0, stream>>>(E, alpha, delta, rr, out);
    }
}
